// GraphConvolutionalNetwork_68255620268813
// MI455X (gfx1250) — compile-verified, matrix operand order checked by reading, not run on hardware
//
#include <hip/hip_runtime.h>
#include <hip/hip_bf16.h>
#include <math.h>

// ---------------------------------------------------------------------------
// GCN forward: 3x (bf16-WMMA GEMM + symmetric-norm edge aggregation + ReLU),
// tri-pool (mean/max/sum) per graph, 3-layer MLP head with sigmoid.
// Node features kept as bf16 in memory so the WMMA GEMM inner loop is pure
// global_load_b128 + v_wmma_f32_16x16x32_bf16 (no in-loop conversion VALU).
// ---------------------------------------------------------------------------

#define N_NODES 50000
#define N_EDGES 800000
#define DFEAT   128
#define N_GRAPHS 512

typedef __attribute__((ext_vector_type(16))) __bf16 v16bf;
typedef __attribute__((ext_vector_type(8)))  float  v8f;
typedef __attribute__((ext_vector_type(8)))  unsigned short us8;

__device__ __forceinline__ unsigned short f2bf_rne(float f) {
    unsigned u = __float_as_uint(f);
    unsigned r = (u + 0x7FFFu + ((u >> 16) & 1u)) >> 16;
    return (unsigned short)r;
}

union Frag { v16bf v; us8 h[2]; unsigned short u[16]; };

// ---- weight prep: W[k][n] (row-major) -> Wt[n][k] bf16 (n-major) ----------
__global__ void wt_convert_kernel(const float* __restrict__ W,
                                  unsigned short* __restrict__ Wt) {
    int t = blockIdx.x * blockDim.x + threadIdx.x;   // 128*128 threads
    if (t >= 128 * 128) return;
    int k = t >> 7, n = t & 127;
    Wt[(size_t)n * 128 + k] = f2bf_rne(W[t]);
}

// ---- input features f32 -> bf16 (row-major) -------------------------------
__global__ void x2bf_kernel(const float* __restrict__ X,
                            unsigned short* __restrict__ Xb, int n4) {
    int t = blockIdx.x * blockDim.x + threadIdx.x;   // one float4 per thread
    if (t >= n4) return;
    float4 v = *(const float4*)(X + (size_t)t * 4);
    ushort4 o;
    o.x = f2bf_rne(v.x); o.y = f2bf_rne(v.y);
    o.z = f2bf_rne(v.z); o.w = f2bf_rne(v.w);
    *(ushort4*)(Xb + (size_t)t * 4) = o;
}

// ---- degree / dinv --------------------------------------------------------
__global__ void deg_kernel(const int* __restrict__ dst, float* __restrict__ deg, int E) {
    int e = blockIdx.x * blockDim.x + threadIdx.x;
    if (e < E) atomicAdd(&deg[dst[e]], 1.0f);
}

__global__ void dinv_kernel(const float* __restrict__ deg, float* __restrict__ dinv, int n) {
    int i = blockIdx.x * blockDim.x + threadIdx.x;
    if (i < n) dinv[i] = rsqrtf(deg[i] + 1.0f);      // +1 self loop
}

// ---- H = Xb(bf16) @ Wt(bf16, n-major), f32 accumulate ---------------------
// grid = 3125 blocks (16 rows each), block = 128 threads = 4 waves.
// Wave w computes output columns [32w, 32w+32) as two 16x16 tiles.
// Fragments per ISA layout: lane = M (A) / N (B) in [0,16); half-wave picks
// K-offset 0/8; vector elements 8..15 hold K+16.
__global__ __launch_bounds__(128)
void gemm_node_kernel(const unsigned short* __restrict__ Xb,
                      const unsigned short* __restrict__ Wt,
                      float* __restrict__ H) {
    const int lane = threadIdx.x & 31;
    const int wave = threadIdx.x >> 5;
    const int rowBase = blockIdx.x << 4;
    const int m  = lane & 15;
    const int kb = (lane >> 4) << 3;     // 0 or 8
    const int n0 = wave << 5;

    v8f acc0 = {};
    v8f acc1 = {};

    const unsigned short* ap  = Xb + (size_t)(rowBase + m) * 128 + kb;
    const unsigned short* bp0 = Wt + (size_t)(n0 + m) * 128 + kb;
    const unsigned short* bp1 = Wt + (size_t)(n0 + 16 + m) * 128 + kb;

#pragma unroll
    for (int kk = 0; kk < 128; kk += 32) {
        Frag a, b0, b1;
        a.h[0]  = *(const us8*)(ap + kk);
        a.h[1]  = *(const us8*)(ap + kk + 16);
        b0.h[0] = *(const us8*)(bp0 + kk);
        b0.h[1] = *(const us8*)(bp0 + kk + 16);
        b1.h[0] = *(const us8*)(bp1 + kk);
        b1.h[1] = *(const us8*)(bp1 + kk + 16);
        acc0 = __builtin_amdgcn_wmma_f32_16x16x32_bf16(
                   false, a.v, false, b0.v, (short)0, acc0, false, false);
        acc1 = __builtin_amdgcn_wmma_f32_16x16x32_bf16(
                   false, a.v, false, b1.v, (short)0, acc1, false, false);
    }

    const int rhalf = (lane >> 4) << 3;  // C/D: lanes 16-31 hold rows M+8
    const int col = n0 + m;
#pragma unroll
    for (int r = 0; r < 8; ++r) {
        const size_t row = (size_t)(rowBase + r + rhalf);
        H[row * DFEAT + col]      = acc0[r];
        H[row * DFEAT + col + 16] = acc1[r];
    }
}

// ---- edge scatter: agg[dst] += h[src] * dinv[src]*dinv[dst] ---------------
// one thread per (edge, 4-float chunk): float4 gather + 4 f32 atomics.
__global__ void edge_agg_kernel(const int* __restrict__ src,
                                const int* __restrict__ dst,
                                const float* __restrict__ dinv,
                                const float* __restrict__ H,
                                float* __restrict__ agg, int E) {
    int t = blockIdx.x * blockDim.x + threadIdx.x;
    int e = t >> 5;
    if (e >= E) return;
    int c = (t & 31) << 2;
    int s = src[e], d = dst[e];
    float norm = dinv[s] * dinv[d];
    float4 v = *(const float4*)(H + (size_t)s * DFEAT + c);
    float* out = agg + (size_t)d * DFEAT + c;
    atomicAdd(out + 0, v.x * norm);
    atomicAdd(out + 1, v.y * norm);
    atomicAdd(out + 2, v.z * norm);
    atomicAdd(out + 3, v.w * norm);
}

// ---- agg = relu(agg + h*dinv^2 + b); also emit bf16 copy for next GEMM ----
__global__ void combine_relu_kernel(float* __restrict__ agg,
                                    const float* __restrict__ H,
                                    const float* __restrict__ dinv,
                                    const float* __restrict__ bias,
                                    unsigned short* __restrict__ outBf, int n) {
    int t = blockIdx.x * blockDim.x + threadIdx.x;
    if (t >= n * DFEAT) return;
    int i = t >> 7, j = t & 127;
    float di = dinv[i];
    float v = agg[t] + H[t] * di * di + bias[j];
    v = v > 0.f ? v : 0.f;
    agg[t] = v;
    outBf[t] = f2bf_rne(v);
}

// ---- tri-pool: sum + max (h>=0 so u32 atomicMax on float bits) + counts ---
__global__ void pool_kernel(const float* __restrict__ H,
                            const int* __restrict__ batch,
                            float* __restrict__ psum,
                            unsigned* __restrict__ pmax,
                            float* __restrict__ counts, int n) {
    int t = blockIdx.x * blockDim.x + threadIdx.x;
    if (t >= n * DFEAT) return;
    int i = t >> 7, j = t & 127;
    int g = batch[i];
    float v = H[t];
    atomicAdd(&psum[(size_t)g * DFEAT + j], v);
    atomicMax(&pmax[(size_t)g * DFEAT + j], __float_as_uint(v));
    if (j == 0) atomicAdd(&counts[g], 1.0f);
}

__global__ void build_g_kernel(const float* __restrict__ psum,
                               const unsigned* __restrict__ pmax,
                               const float* __restrict__ counts,
                               float* __restrict__ g) {
    int t = blockIdx.x * blockDim.x + threadIdx.x;
    if (t >= N_GRAPHS * DFEAT) return;
    int gi = t >> 7, j = t & 127;
    float c = counts[gi];
    float s = psum[t];
    g[(size_t)gi * 384 + j]       = s / fmaxf(c, 1.0f);  // mean
    g[(size_t)gi * 384 + 128 + j] = __uint_as_float(pmax[t]);
    g[(size_t)gi * 384 + 256 + j] = s;                    // sum
}

// ---- small dense layers (fp32, negligible cost) ---------------------------
__global__ void dense_relu_kernel(const float* __restrict__ A,
                                  const float* __restrict__ W,
                                  const float* __restrict__ b,
                                  float* __restrict__ out,
                                  int M, int K, int N) {
    int t = blockIdx.x * blockDim.x + threadIdx.x;
    if (t >= M * N) return;
    int i = t / N, o = t % N;
    float s = b[o];
    const float* a = A + (size_t)i * K;
    for (int k = 0; k < K; ++k) s += a[k] * W[(size_t)k * N + o];
    out[t] = fmaxf(s, 0.f);
}

__global__ void head_sigmoid_kernel(const float* __restrict__ T,
                                    const float* __restrict__ w,
                                    const float* __restrict__ b,
                                    float* __restrict__ out) {
    int gi = blockIdx.x * blockDim.x + threadIdx.x;
    if (gi >= N_GRAPHS) return;
    float s = b[0];
    const float* a = T + (size_t)gi * 128;
    for (int k = 0; k < 128; ++k) s += a[k] * w[k];
    out[gi] = 1.0f / (1.0f + __expf(-s));
}

// ---------------------------------------------------------------------------
extern "C" void kernel_launch(void* const* d_in, const int* in_sizes, int n_in,
                              void* d_out, int out_size, void* d_ws, size_t ws_size,
                              hipStream_t stream) {
    const int N = N_NODES, E = N_EDGES, G = N_GRAPHS;

    const float* x    = (const float*)d_in[0];
    const int*   eidx = (const int*)d_in[1];
    const int*   srcI = eidx;
    const int*   dstI = eidx + E;
    const int*   batch = (const int*)d_in[2];
    const float* W1  = (const float*)d_in[3];
    const float* b1  = (const float*)d_in[4];
    const float* W2  = (const float*)d_in[5];
    const float* b2  = (const float*)d_in[6];
    const float* lw1 = (const float*)d_in[7];
    const float* lb1 = (const float*)d_in[8];
    const float* lw2 = (const float*)d_in[9];
    const float* lb2 = (const float*)d_in[10];
    const float* lw3 = (const float*)d_in[11];
    const float* lb3 = (const float*)d_in[12];
    float* out = (float*)d_out;

    // ---- workspace carve-up ----
    float* bufA = (float*)d_ws;                        // N*128 f32
    float* bufB = bufA + (size_t)N * DFEAT;            // N*128 f32
    float* bufH = bufB + (size_t)N * DFEAT;            // N*128 f32
    unsigned short* xbuf = (unsigned short*)(bufH + (size_t)N * DFEAT); // N*128 bf16
    unsigned short* Wt1 = xbuf + (size_t)N * DFEAT;    // 128*128 bf16
    unsigned short* Wt2 = Wt1 + 128 * 128;
    float* deg   = (float*)(Wt2 + 128 * 128);          // N f32
    float* dinv  = deg + N;                            // N f32
    float* psum  = dinv + N;                           // G*128
    unsigned* pmax = (unsigned*)(psum + (size_t)G * DFEAT); // G*128
    float* counts = (float*)(pmax + (size_t)G * DFEAT);     // G
    float* gbuf  = counts + G;                         // G*384
    float* t1    = gbuf + (size_t)G * 384;             // G*384
    float* t2    = t1 + (size_t)G * 384;               // G*128

    const int TB = 256;
    const int gemmGrid = N / 16;                       // 3125, exact
    const int edgeThreads = E * 32;
    const int nodeThreads = N * DFEAT;

    // weight prep, input conversion, degree/dinv
    wt_convert_kernel<<<(128 * 128 + TB - 1) / TB, TB, 0, stream>>>(W1, Wt1);
    wt_convert_kernel<<<(128 * 128 + TB - 1) / TB, TB, 0, stream>>>(W2, Wt2);
    x2bf_kernel<<<(nodeThreads / 4 + TB - 1) / TB, TB, 0, stream>>>(x, xbuf, nodeThreads / 4);
    hipMemsetAsync(deg, 0, (size_t)N * sizeof(float), stream);
    deg_kernel<<<(E + TB - 1) / TB, TB, 0, stream>>>(dstI, deg, E);
    dinv_kernel<<<(N + TB - 1) / TB, TB, 0, stream>>>(deg, dinv, N);

    // ---- layer 1: x @ W1 ----
    gemm_node_kernel<<<gemmGrid, 128, 0, stream>>>(xbuf, Wt1, bufH);
    hipMemsetAsync(bufA, 0, (size_t)N * DFEAT * sizeof(float), stream);
    edge_agg_kernel<<<(edgeThreads + TB - 1) / TB, TB, 0, stream>>>(srcI, dstI, dinv, bufH, bufA, E);
    combine_relu_kernel<<<(nodeThreads + TB - 1) / TB, TB, 0, stream>>>(bufA, bufH, dinv, b1, xbuf, N);

    // ---- layer 2: h @ W2 ----
    gemm_node_kernel<<<gemmGrid, 128, 0, stream>>>(xbuf, Wt2, bufH);
    hipMemsetAsync(bufB, 0, (size_t)N * DFEAT * sizeof(float), stream);
    edge_agg_kernel<<<(edgeThreads + TB - 1) / TB, TB, 0, stream>>>(srcI, dstI, dinv, bufH, bufB, E);
    combine_relu_kernel<<<(nodeThreads + TB - 1) / TB, TB, 0, stream>>>(bufB, bufH, dinv, b2, xbuf, N);

    // ---- layer 3: h @ W2 (shared weights) ----
    gemm_node_kernel<<<gemmGrid, 128, 0, stream>>>(xbuf, Wt2, bufH);
    hipMemsetAsync(bufA, 0, (size_t)N * DFEAT * sizeof(float), stream);
    edge_agg_kernel<<<(edgeThreads + TB - 1) / TB, TB, 0, stream>>>(srcI, dstI, dinv, bufH, bufA, E);
    combine_relu_kernel<<<(nodeThreads + TB - 1) / TB, TB, 0, stream>>>(bufA, bufH, dinv, b2, xbuf, N);

    // ---- tri-pool ----
    hipMemsetAsync(psum, 0, (size_t)G * DFEAT * sizeof(float), stream);
    hipMemsetAsync(pmax, 0, (size_t)G * DFEAT * sizeof(unsigned), stream);
    hipMemsetAsync(counts, 0, (size_t)G * sizeof(float), stream);
    pool_kernel<<<(nodeThreads + TB - 1) / TB, TB, 0, stream>>>(bufA, batch, psum, pmax, counts, N);
    build_g_kernel<<<(G * DFEAT + TB - 1) / TB, TB, 0, stream>>>(psum, pmax, counts, gbuf);

    // ---- MLP head ----
    dense_relu_kernel<<<(G * 384 + TB - 1) / TB, TB, 0, stream>>>(gbuf, lw1, lb1, t1, G, 384, 384);
    dense_relu_kernel<<<(G * 128 + TB - 1) / TB, TB, 0, stream>>>(t1, lw2, lb2, t2, G, 384, 128);
    head_sigmoid_kernel<<<(G + TB - 1) / TB, TB, 0, stream>>>(t2, lw3, lb3, out);
}
